// ConformerEncoderLayer_19842748907937
// MI455X (gfx1250) — compile-verified, matrix operand order checked by reading, not run on hardware
//
#include <hip/hip_runtime.h>
#include <stdint.h>

typedef __attribute__((ext_vector_type(16))) __bf16 v16bf;
typedef __attribute__((ext_vector_type(8)))  float  v8f;

#define LSEQ   512
#define NBATCH 16
#define DMODEL 512
#define NH     8
#define DHEAD  64
#define FDIM   2048
#define NTOK   (LSEQ * NBATCH)      // 8192 tokens
#define EPSV   1e-5f
#define CHUNK  8                    // heads per attention pass
#define QKVLD  1536

typedef unsigned short ushort_t;

// ---------------------------------------------------------------------------
// helpers
// ---------------------------------------------------------------------------
__device__ __forceinline__ ushort_t f2bf(float f) {
    union { float f; uint32_t u; } c; c.f = f;
    uint32_t u = c.u;
    u += 0x7fffu + ((u >> 16) & 1u);          // round-to-nearest-even
    return (ushort_t)(u >> 16);
}
__device__ __forceinline__ __bf16 bfbits(ushort_t s) {
    return __builtin_bit_cast(__bf16, s);
}
// CDNA5 ISA §7.12.2: 16-bit A 16x32 — lanes 0-15 hold K={0..7,16..23},
// lanes 16-31 hold K={8..15,24..31}; element e (0..15) -> K index.
__device__ __forceinline__ int kmap(int e, int laneHi) {
    int p = e >> 1;
    int base = (p < 4) ? (2 * p) : (2 * p + 8);
    return base + (laneHi ? 8 : 0) + (e & 1);
}
__device__ __forceinline__ float swishf(float v) {
    return v * (1.0f / (1.0f + __expf(-v)));
}
// CDNA5 async copy: 16B global -> LDS, tracked by ASYNCcnt (CDNA5 ISA)
__device__ __forceinline__ void async_cp16(void* lds, const void* gmem) {
    uint32_t l = (uint32_t)(uintptr_t)lds;       // low 32 bits = LDS byte offset
    uint64_t g = (uint64_t)(uintptr_t)gmem;
    asm volatile("global_load_async_to_lds_b128 %0, %1, off"
                 :: "v"(l), "v"(g) : "memory");
}
__device__ __forceinline__ void async_wait0() {
    asm volatile("s_wait_asynccnt 0x0" ::: "memory");
}

// ---------------------------------------------------------------------------
// fp32 -> bf16 cast kernel (weights / pos_emb, once per launch)
// ---------------------------------------------------------------------------
__global__ __launch_bounds__(256) void cast_bf(
        const float* __restrict__ in, ushort_t* __restrict__ out, int n) {
    int i = blockIdx.x * 256 + threadIdx.x;
    if (i < n) out[i] = f2bf(in[i]);
}

// ---------------------------------------------------------------------------
// Generic WMMA GEMM: C[M,N] = act(Abf[M,K] @ Wbf[N,K]^T + bias) (then res + scale*C)
// A,W are bf16.  grid=(N/128, ceil(M/128)); block=256 (8 waves).
// Wave tile 32x64 = 2x4 WMMA.  Async double-buffered LDS staging.
// ---------------------------------------------------------------------------
template<int ACT, bool HAS_BIAS, bool HAS_RES, bool OUT_BF>
__global__ __launch_bounds__(256) void gemm_wmma(
        const ushort_t* __restrict__ A, const ushort_t* __restrict__ W,
        const float* __restrict__ bias, const float* __restrict__ res,
        float res_scale, void* __restrict__ Cout, int M, int K, int N) {
    __shared__ ushort_t As[2][128][32];
    __shared__ ushort_t Bs[2][128][32];
    const int tid    = threadIdx.x;
    const int lane   = tid & 31;
    const int laneHi = (lane >= 16) ? 1 : 0;
    const int wave   = tid >> 5;
    const int wm     = wave & 3;        // 4 M-subtiles of 32
    const int wn     = wave >> 2;       // 2 N-subtiles of 64
    const int m0     = blockIdx.y * 128;
    const int n0     = blockIdx.x * 128;

    v8f acc[2][4];
#pragma unroll
    for (int a = 0; a < 2; ++a)
#pragma unroll
        for (int b = 0; b < 4; ++b)
#pragma unroll
            for (int v = 0; v < 8; ++v) acc[a][b][v] = 0.0f;

    auto stage = [&](int k0, int buf) {
#pragma unroll
        for (int q = 0; q < 2; ++q) {          // A tile: 128 rows x 64B
            int lin = q * 256 + tid;           // 0..511
            int r = lin >> 2, cb = (lin & 3) << 3;
            int gr = m0 + r; if (gr >= M) gr = M - 1;
            async_cp16(&As[buf][r][cb], A + (size_t)gr * K + k0 + cb);
        }
#pragma unroll
        for (int q = 0; q < 2; ++q) {          // W tile: 128 rows x 64B
            int lin = q * 256 + tid;
            int r = lin >> 2, cb = (lin & 3) << 3;
            int gn = n0 + r; if (gn >= N) gn = N - 1;
            async_cp16(&Bs[buf][r][cb], W + (size_t)gn * K + k0 + cb);
        }
    };

    stage(0, 0);
    async_wait0(); __syncthreads();
    int cur = 0;
    for (int k0 = 0; k0 < K; k0 += 32) {
        if (k0 + 32 < K) stage(k0 + 32, cur ^ 1);   // overlap copy with WMMA

        v16bf af[2], bfv[4];
#pragma unroll
        for (int mi = 0; mi < 2; ++mi) {
            int row = wm * 32 + mi * 16 + (lane & 15);
#pragma unroll
            for (int e = 0; e < 16; ++e) af[mi][e] = bfbits(As[cur][row][kmap(e, laneHi)]);
        }
#pragma unroll
        for (int ni = 0; ni < 4; ++ni) {
            int col = wn * 64 + ni * 16 + (lane & 15);
#pragma unroll
            for (int e = 0; e < 16; ++e) bfv[ni][e] = bfbits(Bs[cur][col][kmap(e, laneHi)]);
        }
#pragma unroll
        for (int mi = 0; mi < 2; ++mi)
#pragma unroll
            for (int ni = 0; ni < 4; ++ni)
                acc[mi][ni] = __builtin_amdgcn_wmma_f32_16x16x32_bf16(
                    false, af[mi], false, bfv[ni], (short)0, acc[mi][ni], false, false);

        async_wait0(); __syncthreads();
        cur ^= 1;
    }

#pragma unroll
    for (int mi = 0; mi < 2; ++mi)
#pragma unroll
        for (int ni = 0; ni < 4; ++ni) {
            int colb = n0 + wn * 64 + ni * 16 + (lane & 15);
            float bv = HAS_BIAS ? bias[colb] : 0.0f;
#pragma unroll
            for (int v = 0; v < 8; ++v) {
                int row = m0 + wm * 32 + mi * 16 + v + (laneHi ? 8 : 0);
                if (row < M) {
                    float val = acc[mi][ni][v] + bv;
                    if (ACT == 1) val = swishf(val);
                    size_t idx = (size_t)row * N + colb;
                    if (HAS_RES) val = res[idx] + res_scale * val;
                    if (OUT_BF) ((ushort_t*)Cout)[idx] = f2bf(val);
                    else        ((float*)Cout)[idx]    = val;
                }
            }
        }
}

// ---------------------------------------------------------------------------
// QKV prep: fold bu/bv bias adds, emit bf16 q_u, q_v, k, v (token-major)
// ---------------------------------------------------------------------------
__global__ __launch_bounds__(256) void qkv_prep(
        const float* __restrict__ qkv, const float* __restrict__ bu,
        const float* __restrict__ bv, ushort_t* __restrict__ qu,
        ushort_t* __restrict__ qv, ushort_t* __restrict__ kb,
        ushort_t* __restrict__ vb) {
    int idx = blockIdx.x * 256 + threadIdx.x;   // < NTOK*512
    int t = idx >> 9, c = idx & 511;
    float q = qkv[(size_t)t * QKVLD + c];
    qu[idx] = f2bf(q + bu[c]);
    qv[idx] = f2bf(q + bv[c]);
    kb[idx] = f2bf(qkv[(size_t)t * QKVLD + DMODEL + c]);
    vb[idx] = f2bf(qkv[(size_t)t * QKVLD + 2 * DMODEL + c]);
}

// ---------------------------------------------------------------------------
// Attention AC: score[z][l][s] = sum_d qu[l,d] * k[s,d]   grid=(8,4,CHUNK)
// ---------------------------------------------------------------------------
__global__ __launch_bounds__(256) void attn_ac(
        const ushort_t* __restrict__ qu, const ushort_t* __restrict__ kb,
        float* __restrict__ score, int head_base) {
    __shared__ ushort_t As[2][128][32];
    __shared__ ushort_t Bs[2][64][32];
    const int z = blockIdx.z, g = head_base + z;
    const int n = g / NH, h = g % NH;
    const int tid = threadIdx.x, lane = tid & 31, laneHi = (lane >= 16) ? 1 : 0;
    const int wave = tid >> 5, wm = wave & 3, wn = wave >> 2;
    const int m0 = blockIdx.y * 128, n0 = blockIdx.x * 64;

    v8f acc[2][2];
#pragma unroll
    for (int a = 0; a < 2; ++a)
#pragma unroll
        for (int b = 0; b < 2; ++b)
#pragma unroll
            for (int v = 0; v < 8; ++v) acc[a][b][v] = 0.0f;

    auto stage = [&](int k0, int buf) {
#pragma unroll
        for (int q = 0; q < 2; ++q) {
            int lin = q * 256 + tid, r = lin >> 2, cb = (lin & 3) << 3;
            async_cp16(&As[buf][r][cb],
                qu + ((size_t)(m0 + r) * NBATCH + n) * DMODEL + h * DHEAD + k0 + cb);
        }
        {   int r = tid >> 2, cb = (tid & 3) << 3;
            async_cp16(&Bs[buf][r][cb],
                kb + ((size_t)(n0 + r) * NBATCH + n) * DMODEL + h * DHEAD + k0 + cb);
        }
    };

    stage(0, 0);
    async_wait0(); __syncthreads();
    int cur = 0;
    for (int k0 = 0; k0 < DHEAD; k0 += 32) {
        if (k0 + 32 < DHEAD) stage(k0 + 32, cur ^ 1);
        v16bf af[2], bfv[2];
#pragma unroll
        for (int mi = 0; mi < 2; ++mi) {
            int row = wm * 32 + mi * 16 + (lane & 15);
#pragma unroll
            for (int e = 0; e < 16; ++e) af[mi][e] = bfbits(As[cur][row][kmap(e, laneHi)]);
        }
#pragma unroll
        for (int ni = 0; ni < 2; ++ni) {
            int col = wn * 32 + ni * 16 + (lane & 15);
#pragma unroll
            for (int e = 0; e < 16; ++e) bfv[ni][e] = bfbits(Bs[cur][col][kmap(e, laneHi)]);
        }
#pragma unroll
        for (int mi = 0; mi < 2; ++mi)
#pragma unroll
            for (int ni = 0; ni < 2; ++ni)
                acc[mi][ni] = __builtin_amdgcn_wmma_f32_16x16x32_bf16(
                    false, af[mi], false, bfv[ni], (short)0, acc[mi][ni], false, false);
        async_wait0(); __syncthreads();
        cur ^= 1;
    }
#pragma unroll
    for (int mi = 0; mi < 2; ++mi)
#pragma unroll
        for (int ni = 0; ni < 2; ++ni) {
            int col = n0 + wn * 32 + ni * 16 + (lane & 15);
#pragma unroll
            for (int v = 0; v < 8; ++v) {
                int row = m0 + wm * 32 + mi * 16 + v + (laneHi ? 8 : 0);
                score[((size_t)z * LSEQ + row) * LSEQ + col] = acc[mi][ni][v];
            }
        }
}

// ---------------------------------------------------------------------------
// Attention BD: bd[z][l][m] = sum_d qv[l,d] * p[m, h*64+d]   grid=(16,4,CHUNK)
// ---------------------------------------------------------------------------
__global__ __launch_bounds__(256) void attn_bd(
        const ushort_t* __restrict__ qv, const ushort_t* __restrict__ pbf,
        float* __restrict__ bd, int head_base) {
    __shared__ ushort_t As[2][128][32];
    __shared__ ushort_t Bs[2][64][32];
    const int z = blockIdx.z, g = head_base + z;
    const int n = g / NH, h = g % NH;
    const int tid = threadIdx.x, lane = tid & 31, laneHi = (lane >= 16) ? 1 : 0;
    const int wave = tid >> 5, wm = wave & 3, wn = wave >> 2;
    const int m0 = blockIdx.y * 128, n0 = blockIdx.x * 64;

    v8f acc[2][2];
#pragma unroll
    for (int a = 0; a < 2; ++a)
#pragma unroll
        for (int b = 0; b < 2; ++b)
#pragma unroll
            for (int v = 0; v < 8; ++v) acc[a][b][v] = 0.0f;

    auto stage = [&](int k0, int buf) {
#pragma unroll
        for (int q = 0; q < 2; ++q) {
            int lin = q * 256 + tid, r = lin >> 2, cb = (lin & 3) << 3;
            async_cp16(&As[buf][r][cb],
                qv + ((size_t)(m0 + r) * NBATCH + n) * DMODEL + h * DHEAD + k0 + cb);
        }
        {   int r = tid >> 2, cb = (tid & 3) << 3;
            int m = n0 + r; if (m > 2 * LSEQ - 2) m = 2 * LSEQ - 2;  // clamp, col 1023 never read
            async_cp16(&Bs[buf][r][cb],
                pbf + (size_t)m * DMODEL + h * DHEAD + k0 + cb);
        }
    };

    stage(0, 0);
    async_wait0(); __syncthreads();
    int cur = 0;
    for (int k0 = 0; k0 < DHEAD; k0 += 32) {
        if (k0 + 32 < DHEAD) stage(k0 + 32, cur ^ 1);
        v16bf af[2], bfv[2];
#pragma unroll
        for (int mi = 0; mi < 2; ++mi) {
            int row = wm * 32 + mi * 16 + (lane & 15);
#pragma unroll
            for (int e = 0; e < 16; ++e) af[mi][e] = bfbits(As[cur][row][kmap(e, laneHi)]);
        }
#pragma unroll
        for (int ni = 0; ni < 2; ++ni) {
            int col = wn * 32 + ni * 16 + (lane & 15);
#pragma unroll
            for (int e = 0; e < 16; ++e) bfv[ni][e] = bfbits(Bs[cur][col][kmap(e, laneHi)]);
        }
#pragma unroll
        for (int mi = 0; mi < 2; ++mi)
#pragma unroll
            for (int ni = 0; ni < 2; ++ni)
                acc[mi][ni] = __builtin_amdgcn_wmma_f32_16x16x32_bf16(
                    false, af[mi], false, bfv[ni], (short)0, acc[mi][ni], false, false);
        async_wait0(); __syncthreads();
        cur ^= 1;
    }
#pragma unroll
    for (int mi = 0; mi < 2; ++mi)
#pragma unroll
        for (int ni = 0; ni < 2; ++ni) {
            int col = n0 + wn * 32 + ni * 16 + (lane & 15);   // <=1023, stride 1024
#pragma unroll
            for (int v = 0; v < 8; ++v) {
                int row = m0 + wm * 32 + mi * 16 + v + (laneHi ? 8 : 0);
                bd[((size_t)z * LSEQ + row) * 1024 + col] = acc[mi][ni][v];
            }
        }
}

// ---------------------------------------------------------------------------
// fused rel_shift + softmax -> bf16 probs.  grid=(LSEQ, CHUNK), block=128
// ---------------------------------------------------------------------------
__global__ __launch_bounds__(128) void attn_softmax(
        const float* __restrict__ score, const float* __restrict__ bd,
        ushort_t* __restrict__ prob) {
    const int l = blockIdx.x, z = blockIdx.y, t = threadIdx.x;
    const float* row  = score + ((size_t)z * LSEQ + l) * LSEQ;
    const float* brow = bd + ((size_t)z * LSEQ + l) * 1024;
    ushort_t* orow = prob + ((size_t)z * LSEQ + l) * LSEQ;
    __shared__ float red[128];
    float vals[4]; float mx = -3.4e38f;
#pragma unroll
    for (int i = 0; i < 4; ++i) {
        int s = i * 128 + t;
        float v = (row[s] + brow[(LSEQ - 1) - l + s]) * 0.125f;
        vals[i] = v; mx = fmaxf(mx, v);
    }
    red[t] = mx; __syncthreads();
    for (int o = 64; o > 0; o >>= 1) { if (t < o) red[t] = fmaxf(red[t], red[t + o]); __syncthreads(); }
    mx = red[0]; __syncthreads();
    float sum = 0.f;
#pragma unroll
    for (int i = 0; i < 4; ++i) { vals[i] = __expf(vals[i] - mx); sum += vals[i]; }
    red[t] = sum; __syncthreads();
    for (int o = 64; o > 0; o >>= 1) { if (t < o) red[t] += red[t + o]; __syncthreads(); }
    float inv = 1.0f / red[0];
#pragma unroll
    for (int i = 0; i < 4; ++i) orow[i * 128 + t] = f2bf(vals[i] * inv);
}

// ---------------------------------------------------------------------------
// attn @ V -> bf16 out[l,n,h*64+d].  K=512, N=64.  grid=(1,4,CHUNK)
// ---------------------------------------------------------------------------
__global__ __launch_bounds__(256) void attn_av(
        const ushort_t* __restrict__ prob, const ushort_t* __restrict__ vb,
        ushort_t* __restrict__ out, int head_base) {
    __shared__ ushort_t As[2][128][32];
    __shared__ ushort_t Vs[2][32][64];   // [k][d] : B is K-major
    const int z = blockIdx.z, g = head_base + z;
    const int n = g / NH, h = g % NH;
    const int tid = threadIdx.x, lane = tid & 31, laneHi = (lane >= 16) ? 1 : 0;
    const int wave = tid >> 5, wm = wave & 3, wn = wave >> 2;   // wn in {0,1}
    const int m0 = blockIdx.y * 128;

    v8f acc[2][2];
#pragma unroll
    for (int a = 0; a < 2; ++a)
#pragma unroll
        for (int b = 0; b < 2; ++b)
#pragma unroll
            for (int v = 0; v < 8; ++v) acc[a][b][v] = 0.0f;

    auto stage = [&](int k0, int buf) {
#pragma unroll
        for (int q = 0; q < 2; ++q) {
            int lin = q * 256 + tid, r = lin >> 2, cb = (lin & 3) << 3;
            async_cp16(&As[buf][r][cb],
                prob + ((size_t)z * LSEQ + m0 + r) * LSEQ + k0 + cb);
        }
        {   int r = tid >> 3, cb = (tid & 7) << 3;
            async_cp16(&Vs[buf][r][cb],
                vb + ((size_t)(k0 + r) * NBATCH + n) * DMODEL + h * DHEAD + cb);
        }
    };

    stage(0, 0);
    async_wait0(); __syncthreads();
    int cur = 0;
    for (int k0 = 0; k0 < LSEQ; k0 += 32) {
        if (k0 + 32 < LSEQ) stage(k0 + 32, cur ^ 1);
        v16bf af[2], bfv[2];
#pragma unroll
        for (int mi = 0; mi < 2; ++mi) {
            int row = wm * 32 + mi * 16 + (lane & 15);
#pragma unroll
            for (int e = 0; e < 16; ++e) af[mi][e] = bfbits(As[cur][row][kmap(e, laneHi)]);
        }
#pragma unroll
        for (int ni = 0; ni < 2; ++ni) {
            int col = wn * 32 + ni * 16 + (lane & 15);
#pragma unroll
            for (int e = 0; e < 16; ++e) bfv[ni][e] = bfbits(Vs[cur][kmap(e, laneHi)][col]);
        }
#pragma unroll
        for (int mi = 0; mi < 2; ++mi)
#pragma unroll
            for (int ni = 0; ni < 2; ++ni)
                acc[mi][ni] = __builtin_amdgcn_wmma_f32_16x16x32_bf16(
                    false, af[mi], false, bfv[ni], (short)0, acc[mi][ni], false, false);
        async_wait0(); __syncthreads();
        cur ^= 1;
    }
#pragma unroll
    for (int mi = 0; mi < 2; ++mi)
#pragma unroll
        for (int ni = 0; ni < 2; ++ni) {
            int d = wn * 32 + ni * 16 + (lane & 15);
#pragma unroll
            for (int v = 0; v < 8; ++v) {
                int l = m0 + wm * 32 + mi * 16 + v + (laneHi ? 8 : 0);
                out[((size_t)l * NBATCH + n) * DMODEL + h * DHEAD + d] = f2bf(acc[mi][ni][v]);
            }
        }
}

// ---------------------------------------------------------------------------
// LayerNorm (512) -> bf16 (feeds GEMMs) or fp32 (final). block=128 per row
// ---------------------------------------------------------------------------
template<bool OUTBF>
__global__ __launch_bounds__(128) void layernorm_k(
        const float* __restrict__ x, const float* __restrict__ g,
        const float* __restrict__ b, void* __restrict__ outp) {
    const int r = blockIdx.x, t = threadIdx.x;
    const float* xr = x + (size_t)r * DMODEL;
    __shared__ float red[128];
    float v[4]; float s = 0.f;
#pragma unroll
    for (int i = 0; i < 4; ++i) { v[i] = xr[i * 128 + t]; s += v[i]; }
    red[t] = s; __syncthreads();
    for (int o = 64; o > 0; o >>= 1) { if (t < o) red[t] += red[t + o]; __syncthreads(); }
    float mean = red[0] * (1.0f / DMODEL); __syncthreads();
    float s2 = 0.f;
#pragma unroll
    for (int i = 0; i < 4; ++i) { float d = v[i] - mean; s2 += d * d; }
    red[t] = s2; __syncthreads();
    for (int o = 64; o > 0; o >>= 1) { if (t < o) red[t] += red[t + o]; __syncthreads(); }
    float rstd = rsqrtf(red[0] * (1.0f / DMODEL) + EPSV);
#pragma unroll
    for (int i = 0; i < 4; ++i) {
        int c = i * 128 + t;
        float y = (v[i] - mean) * rstd * g[c] + b[c];
        if (OUTBF) ((ushort_t*)outp)[(size_t)r * DMODEL + c] = f2bf(y);
        else       ((float*)outp)[(size_t)r * DMODEL + c]    = y;
    }
}

// ---------------------------------------------------------------------------
// GLU: out[t,c] = a * sigmoid(gate), in [tok,1024] fp32 -> fp32
// ---------------------------------------------------------------------------
__global__ __launch_bounds__(256) void glu_k(
        const float* __restrict__ in, float* __restrict__ out) {
    int idx = blockIdx.x * 256 + threadIdx.x;   // < NTOK*512
    int t = idx >> 9, c = idx & 511;
    float a  = in[(size_t)t * 1024 + c];
    float gt = in[(size_t)t * 1024 + 512 + c];
    out[idx] = a * (1.0f / (1.0f + __expf(-gt)));
}

// ---------------------------------------------------------------------------
// Depthwise conv (K=31, pad 15) + BatchNorm(eval) + swish -> bf16
// ---------------------------------------------------------------------------
__global__ __launch_bounds__(256) void dwconv_bn_swish(
        const float* __restrict__ in, const float* __restrict__ w,
        const float* __restrict__ bias, const float* __restrict__ gamma,
        const float* __restrict__ beta, const float* __restrict__ mean,
        const float* __restrict__ var, ushort_t* __restrict__ out) {
    int idx = blockIdx.x * 256 + threadIdx.x;   // < NTOK*512
    int c = idx & 511, t = idx >> 9;
    int n = t & (NBATCH - 1), l = t >> 4;
    float acc = bias[c];
#pragma unroll
    for (int j = 0; j < 31; ++j) {
        int lp = l + j - 15;
        if (lp >= 0 && lp < LSEQ)
            acc += in[(((size_t)lp * NBATCH + n) << 9) + c] * w[c * 31 + j];
    }
    float y = (acc - mean[c]) * rsqrtf(var[c] + EPSV) * gamma[c] + beta[c];
    out[idx] = f2bf(swishf(y));
}

// ---------------------------------------------------------------------------
// host-side orchestration
// ---------------------------------------------------------------------------
extern "C" void kernel_launch(void* const* d_in, const int* in_sizes, int n_in,
                              void* d_out, int out_size, void* d_ws, size_t ws_size,
                              hipStream_t stream) {
    const float* src     = (const float*)d_in[0];
    const float* pos_emb = (const float*)d_in[1];
    const float* ffm_w1  = (const float*)d_in[2];
    const float* ffm_b1  = (const float*)d_in[3];
    const float* ffm_w2  = (const float*)d_in[4];
    const float* ffm_b2  = (const float*)d_in[5];
    const float* ff_w1   = (const float*)d_in[6];
    const float* ff_b1   = (const float*)d_in[7];
    const float* ff_w2   = (const float*)d_in[8];
    const float* ff_b2   = (const float*)d_in[9];
    const float* in_w    = (const float*)d_in[10];
    const float* in_b    = (const float*)d_in[11];
    const float* out_w   = (const float*)d_in[12];
    const float* out_b   = (const float*)d_in[13];
    const float* pos_w   = (const float*)d_in[14];
    const float* bu      = (const float*)d_in[15];
    const float* bv      = (const float*)d_in[16];
    const float* pw1_w   = (const float*)d_in[17];
    const float* pw1_b   = (const float*)d_in[18];
    const float* dw_w    = (const float*)d_in[19];
    const float* dw_b    = (const float*)d_in[20];
    const float* bn_g    = (const float*)d_in[21];
    const float* bn_b    = (const float*)d_in[22];
    const float* bn_m    = (const float*)d_in[23];
    const float* bn_v    = (const float*)d_in[24];
    const float* pw2_w   = (const float*)d_in[25];
    const float* pw2_b   = (const float*)d_in[26];
    const float* ln_ffm_g = (const float*)d_in[27];
    const float* ln_ffm_b = (const float*)d_in[28];
    const float* ln_mha_g = (const float*)d_in[29];
    const float* ln_mha_b = (const float*)d_in[30];
    const float* ln_conv_g = (const float*)d_in[31];
    const float* ln_conv_b = (const float*)d_in[32];
    const float* ln_ff_g  = (const float*)d_in[33];
    const float* ln_ff_b  = (const float*)d_in[34];
    const float* ln_fin_g = (const float*)d_in[35];
    const float* ln_fin_b = (const float*)d_in[36];

    char* ws = (char*)d_ws;
    size_t off = 0;
    auto carveB = [&](size_t bytes) -> void* {
        void* p = (void*)(ws + off);
        off += (bytes + 255) & ~(size_t)255;
        return p;
    };
    // fp32 buffers
    float* bufX1  = (float*)carveB((size_t)NTOK * DMODEL * 4);
    float* bufX2  = (float*)carveB((size_t)NTOK * DMODEL * 4);
    float* bufQKV = (float*)carveB((size_t)NTOK * QKVLD * 4);   // reused for pw1/GLU
    float* bufS   = (float*)carveB((size_t)CHUNK * LSEQ * LSEQ * 4);
    float* bufBD  = (float*)carveB((size_t)CHUNK * LSEQ * 1024 * 4);
    // bf16 buffers
    ushort_t* lnbf   = (ushort_t*)carveB((size_t)NTOK * DMODEL * 2);
    ushort_t* fbf    = (ushort_t*)carveB((size_t)NTOK * FDIM * 2);
    ushort_t* qubf   = (ushort_t*)carveB((size_t)NTOK * DMODEL * 2);
    ushort_t* qvbf   = (ushort_t*)carveB((size_t)NTOK * DMODEL * 2);
    ushort_t* kbf    = (ushort_t*)carveB((size_t)NTOK * DMODEL * 2);
    ushort_t* vbf    = (ushort_t*)carveB((size_t)NTOK * DMODEL * 2);
    ushort_t* probbf = (ushort_t*)carveB((size_t)CHUNK * LSEQ * LSEQ * 2);
    ushort_t* aobf   = (ushort_t*)carveB((size_t)NTOK * DMODEL * 2);
    ushort_t* dwbf   = (ushort_t*)carveB((size_t)NTOK * DMODEL * 2);
    ushort_t* pbf    = (ushort_t*)carveB((size_t)1024 * DMODEL * 2);
    ushort_t* pinbf  = (ushort_t*)carveB((size_t)1024 * DMODEL * 2);
    // bf16 weights
    ushort_t* wffm1 = (ushort_t*)carveB((size_t)FDIM * DMODEL * 2);
    ushort_t* wffm2 = (ushort_t*)carveB((size_t)DMODEL * FDIM * 2);
    ushort_t* wff1  = (ushort_t*)carveB((size_t)FDIM * DMODEL * 2);
    ushort_t* wff2  = (ushort_t*)carveB((size_t)DMODEL * FDIM * 2);
    ushort_t* win   = (ushort_t*)carveB((size_t)QKVLD * DMODEL * 2);
    ushort_t* wout  = (ushort_t*)carveB((size_t)DMODEL * DMODEL * 2);
    ushort_t* wpos  = (ushort_t*)carveB((size_t)DMODEL * DMODEL * 2);
    ushort_t* wpw1  = (ushort_t*)carveB((size_t)1024 * DMODEL * 2);
    ushort_t* wpw2  = (ushort_t*)carveB((size_t)DMODEL * DMODEL * 2);
    // conv fp32 temporaries overlay freed QKV fp32 region
    float* pw1out = bufQKV;                                  // NTOK*1024 floats
    float* bufG   = bufQKV + (size_t)NTOK * 1024;            // NTOK*512 floats

    auto castw = [&](const float* s, ushort_t* d, int n) {
        cast_bf<<<(n + 255) / 256, 256, 0, stream>>>(s, d, n);
    };
    castw(ffm_w1, wffm1, FDIM * DMODEL);   castw(ffm_w2, wffm2, DMODEL * FDIM);
    castw(ff_w1,  wff1,  FDIM * DMODEL);   castw(ff_w2,  wff2,  DMODEL * FDIM);
    castw(in_w,   win,   QKVLD * DMODEL);  castw(out_w,  wout,  DMODEL * DMODEL);
    castw(pos_w,  wpos,  DMODEL * DMODEL); castw(pw1_w,  wpw1,  1024 * DMODEL);
    castw(pw2_w,  wpw2,  DMODEL * DMODEL);
    castw(pos_emb, pinbf, (2 * LSEQ - 1) * DMODEL);

    // ---- FFN macaron 1: x1 = src + 0.5*ffn(LN(src)) ----
    layernorm_k<true><<<NTOK, 128, 0, stream>>>(src, ln_ffm_g, ln_ffm_b, lnbf);
    gemm_wmma<1, true, false, true><<<dim3(FDIM / 128, NTOK / 128), 256, 0, stream>>>(
        lnbf, wffm1, ffm_b1, nullptr, 0.f, fbf, NTOK, DMODEL, FDIM);
    gemm_wmma<0, true, true, false><<<dim3(DMODEL / 128, NTOK / 128), 256, 0, stream>>>(
        fbf, wffm2, ffm_b2, src, 0.5f, bufX1, NTOK, FDIM, DMODEL);

    // ---- Attention: x2 = x1 + rel_attn(LN(x1)) ----
    layernorm_k<true><<<NTOK, 128, 0, stream>>>(bufX1, ln_mha_g, ln_mha_b, lnbf);
    gemm_wmma<0, true, false, false><<<dim3(QKVLD / 128, NTOK / 128), 256, 0, stream>>>(
        lnbf, win, in_b, nullptr, 0.f, bufQKV, NTOK, DMODEL, QKVLD);
    gemm_wmma<0, false, false, true><<<dim3(DMODEL / 128, 8), 256, 0, stream>>>(
        pinbf, wpos, nullptr, nullptr, 0.f, pbf, 2 * LSEQ - 1, DMODEL, DMODEL);
    qkv_prep<<<(NTOK * DMODEL) / 256, 256, 0, stream>>>(bufQKV, bu, bv, qubf, qvbf, kbf, vbf);

    for (int hb = 0; hb < NBATCH * NH; hb += CHUNK) {
        attn_ac<<<dim3(LSEQ / 64, LSEQ / 128, CHUNK), 256, 0, stream>>>(qubf, kbf, bufS, hb);
        attn_bd<<<dim3(1024 / 64, LSEQ / 128, CHUNK), 256, 0, stream>>>(qvbf, pbf, bufBD, hb);
        attn_softmax<<<dim3(LSEQ, CHUNK), 128, 0, stream>>>(bufS, bufBD, probbf);
        attn_av<<<dim3(1, LSEQ / 128, CHUNK), 256, 0, stream>>>(probbf, vbf, aobf, hb);
    }
    gemm_wmma<0, true, true, false><<<dim3(DMODEL / 128, NTOK / 128), 256, 0, stream>>>(
        aobf, wout, out_b, bufX1, 1.0f, bufX2, NTOK, DMODEL, DMODEL);

    // ---- Conv module: x3 = x2 + conv(LN(x2)) ----
    layernorm_k<true><<<NTOK, 128, 0, stream>>>(bufX2, ln_conv_g, ln_conv_b, lnbf);
    gemm_wmma<0, true, false, false><<<dim3(1024 / 128, NTOK / 128), 256, 0, stream>>>(
        lnbf, wpw1, pw1_b, nullptr, 0.f, pw1out, NTOK, DMODEL, 1024);
    glu_k<<<(NTOK * DMODEL) / 256, 256, 0, stream>>>(pw1out, bufG);
    dwconv_bn_swish<<<(NTOK * DMODEL) / 256, 256, 0, stream>>>(
        bufG, dw_w, dw_b, bn_g, bn_b, bn_m, bn_v, dwbf);
    gemm_wmma<0, true, true, false><<<dim3(DMODEL / 128, NTOK / 128), 256, 0, stream>>>(
        dwbf, wpw2, pw2_b, bufX2, 1.0f, bufX1, NTOK, DMODEL, DMODEL);

    // ---- FFN macaron 2: x4 = x3 + 0.5*ffn(LN(x3)) ----
    layernorm_k<true><<<NTOK, 128, 0, stream>>>(bufX1, ln_ff_g, ln_ff_b, lnbf);
    gemm_wmma<1, true, false, true><<<dim3(FDIM / 128, NTOK / 128), 256, 0, stream>>>(
        lnbf, wff1, ff_b1, nullptr, 0.f, fbf, NTOK, DMODEL, FDIM);
    gemm_wmma<0, true, true, false><<<dim3(DMODEL / 128, NTOK / 128), 256, 0, stream>>>(
        fbf, wff2, ff_b2, bufX1, 0.5f, bufX2, NTOK, FDIM, DMODEL);

    // ---- final LayerNorm -> d_out ----
    layernorm_k<false><<<NTOK, 128, 0, stream>>>(bufX2, ln_fin_g, ln_fin_b, (float*)d_out);
}
